// AttentionBlock_14723147891282
// MI455X (gfx1250) — matrix operand order repaired and checked by reading, not compiled, not run on hardware
//
#include <hip/hip_runtime.h>
#include <hip/hip_bf16.h>

typedef _Float16 half_t;
typedef half_t v16h __attribute__((ext_vector_type(16)));
typedef half_t v8h  __attribute__((ext_vector_type(8)));
typedef float  v8f  __attribute__((ext_vector_type(8)));

constexpr int Bz = 8;      // batch
constexpr int Lz = 1024;   // seq len
constexpr int NC = 512;    // channels
constexpr int Gz = 32;     // groupnorm groups
constexpr int Hz = 8;      // heads
constexpr int DH = 64;     // head dim
constexpr float EPSF  = 1e-5f;
constexpr float SCALE = 0.125f; // dh^-0.5

// ---------------- WMMA helpers (CDNA5 v_wmma_f32_16x16x32_f16) ----------------

__device__ __forceinline__ v8f wmma_f16(v16h a, v16h b, v8f c) {
    // 8 args: (neg_a, A, neg_b, B, c_mod, C, reuse_a, reuse_b)
    return __builtin_amdgcn_wmma_f32_16x16x32_f16(false, a, false, b, (short)0, c, false, false);
}

// A fragment: 16(M) x 32(K), row-major source, lda elements.
// ISA layout: lanes 0-15 M=lane, lanes 16-31 M=lane-16;
//   VGPR0-3: K = 8*half + {0..7}; VGPR4-7: K = 16 + 8*half + {0..7}.
__device__ __forceinline__ v16h ld_afrag(const half_t* base, int lda, int k0, int lane) {
    int m = lane & 15, hf = lane >> 4;
    const half_t* p = base + (size_t)m * lda + k0 + 8 * hf;
    v8h lo = *(const v8h*)(p);
    v8h hi = *(const v8h*)(p + 16);
    v16h r;
#pragma unroll
    for (int i = 0; i < 8; ++i) { r[i] = lo[i]; r[i + 8] = hi[i]; }
    return r;
}

// B fragment: 32(K) x 16(N), source stored transposed as bt[N][ldb>=K] (element (k,n) at bt[n*ldb+k]).
// ISA layout: lane n = lane&15; VGPR0-3: K = 8*half + {0..7}; VGPR4-7: K = 16 + 8*half + {0..7}.
__device__ __forceinline__ v16h ld_bfrag(const half_t* bt, int ldb, int k0, int n0, int lane) {
    int n = lane & 15, hf = lane >> 4;
    const half_t* p = bt + (size_t)(n0 + n) * ldb + k0 + 8 * hf;
    v8h lo = *(const v8h*)(p);
    v8h hi = *(const v8h*)(p + 16);
    v16h r;
#pragma unroll
    for (int i = 0; i < 8; ++i) { r[i] = lo[i]; r[i + 8] = hi[i]; }
    return r;
}

// A fragment from an LDS 16x32 f16 tile (row stride 32).
__device__ __forceinline__ v16h ld_afrag_lds(const half_t* p, int lane) {
    int m = lane & 15, hf = lane >> 4;
    const half_t* row = p + m * 32 + 8 * hf;
    v16h r;
#pragma unroll
    for (int i = 0; i < 8; ++i) { r[i] = row[i]; r[i + 8] = row[16 + i]; }
    return r;
}

__device__ __forceinline__ float rowmax16(float v) {
    v = fmaxf(v, __shfl_xor(v, 1, 32));
    v = fmaxf(v, __shfl_xor(v, 2, 32));
    v = fmaxf(v, __shfl_xor(v, 4, 32));
    v = fmaxf(v, __shfl_xor(v, 8, 32));
    return v;
}
__device__ __forceinline__ float rowsum16(float v) {
    v += __shfl_xor(v, 1, 32);
    v += __shfl_xor(v, 2, 32);
    v += __shfl_xor(v, 4, 32);
    v += __shfl_xor(v, 8, 32);
    return v;
}

// ---------------- Kernel 1: fp32 [K][N] weight -> f16 transposed [N][K] ----------------

__global__ void w_transpose_f16_kernel(const float* __restrict__ W, half_t* __restrict__ WT,
                                       int K, int N) {
    int idx = blockIdx.x * blockDim.x + threadIdx.x;
    if (idx < K * N) {
        int k = idx / N, n = idx % N;
        WT[(size_t)n * K + k] = (half_t)W[idx];
    }
}

// ---------------- Kernel 2: masked GroupNorm -> f16 xn (padded rows zeroed) ----------------

__global__ void groupnorm_kernel(const float* __restrict__ x, const int* __restrict__ lengths,
                                 const float* __restrict__ gamma, const float* __restrict__ beta,
                                 half_t* __restrict__ xn) {
    constexpr int CPG = NC / Gz; // 16
    int b = blockIdx.x / Gz, g = blockIdx.x % Gz;
    int len = lengths[b];
    int t  = threadIdx.x;        // 256 threads
    int cl = t & (CPG - 1);      // channel in group
    int lr = t >> 4;             // 16 row-lanes
    int c  = g * CPG + cl;
    const float* xb = x + (size_t)b * Lz * NC;

    float s = 0.f, ss = 0.f;
    for (int l = lr; l < len; l += 16) {
        float v = xb[(size_t)l * NC + c];
        s += v; ss += v * v;
    }
    __shared__ float rs[256], rss[256];
    rs[t] = s; rss[t] = ss;
    __syncthreads();
    for (int off = 128; off > 0; off >>= 1) {
        if (t < off) { rs[t] += rs[t + off]; rss[t] += rss[t + off]; }
        __syncthreads();
    }
    __shared__ float smean, srstd;
    if (t == 0) {
        float cnt  = (float)len * (float)CPG;
        float mean = rs[0] / cnt;
        float var  = rss[0] / cnt - mean * mean;
        smean = mean;
        srstd = rsqrtf(var + EPSF);
    }
    __syncthreads();
    float mean = smean, rstd = srstd, gm = gamma[c], bt = beta[c];
    for (int l = lr; l < Lz; l += 16) {
        float v = (l < len) ? (xb[(size_t)l * NC + c] - mean) * rstd * gm + bt : 0.f;
        xn[((size_t)b * Lz + l) * NC + c] = (half_t)v;
    }
}

// ---------------- Kernel 3: QKV GEMM (WMMA, 64x32 per wave) + scatter to q / k / vT ----------------
// Grid: (M/128, 3C/128) = (64, 12); block = 8 waves arranged 2(M) x 4(N).

__global__ void qkv_gemm_kernel(const half_t* __restrict__ xn, const half_t* __restrict__ WqkvT,
                                const float* __restrict__ bqkv,
                                half_t* __restrict__ qb, half_t* __restrict__ kbuf,
                                half_t* __restrict__ vtb) {
    int lane = threadIdx.x & 31;
    int wv   = threadIdx.x >> 5;
    int mw = wv >> 2, nw = wv & 3;
    int m0 = blockIdx.x * 128 + mw * 64;   // 64-row strip, b-aligned (1024 % 64 == 0)
    int n0 = blockIdx.y * 128 + nw * 32;

    v8f acc[4][2] = {};
    for (int k0 = 0; k0 < NC; k0 += 32) {
        v16h a[4], bf[2];
#pragma unroll
        for (int mi = 0; mi < 4; ++mi)
            a[mi] = ld_afrag(xn + (size_t)(m0 + 16 * mi) * NC, NC, k0, lane);
#pragma unroll
        for (int ni = 0; ni < 2; ++ni)
            bf[ni] = ld_bfrag(WqkvT, NC, k0, n0 + 16 * ni, lane);
#pragma unroll
        for (int mi = 0; mi < 4; ++mi)
#pragma unroll
            for (int ni = 0; ni < 2; ++ni)
                acc[mi][ni] = wmma_f16(a[mi], bf[ni], acc[mi][ni]);
    }

    int n = lane & 15, hf = lane >> 4;
    int bi = m0 >> 10;                  // batch index (constant per wave)
    int lbase = (m0 & 1023) + 8 * hf;
#pragma unroll
    for (int ni = 0; ni < 2; ++ni) {
        int col = n0 + 16 * ni + n;
        int which = col >> 9;           // 0:q 1:k 2:v (uniform per wave subtile)
        int c = col & 511;
        int h = c >> 6, d = c & 63;
        float bias = bqkv[col];
        size_t bh = (size_t)bi * Hz + h;
        if (which == 2) {
            // vT [B,H,64,L]: contiguous in l per lane
            half_t* p = vtb + (bh * DH + d) * Lz + lbase;
#pragma unroll
            for (int mi = 0; mi < 4; ++mi)
#pragma unroll
                for (int r = 0; r < 8; ++r)
                    p[16 * mi + r] = (half_t)(acc[mi][ni][r] + bias);
        } else {
            half_t* p = ((which == 0) ? qb : kbuf) + bh * Lz * DH + d + (size_t)lbase * DH;
#pragma unroll
            for (int mi = 0; mi < 4; ++mi)
#pragma unroll
                for (int r = 0; r < 8; ++r)
                    p[(16 * mi + r) * DH] = (half_t)(acc[mi][ni][r] + bias);
        }
    }
}

// ---------------- Kernel 4: per-(b,h) flash attention, 8 waves x 16 queries ----------------

__global__ void attention_kernel(const half_t* __restrict__ qb, const half_t* __restrict__ kbuf,
                                 const half_t* __restrict__ vtb, const int* __restrict__ lengths,
                                 half_t* __restrict__ attn) {
    constexpr int QB = Lz / 128;  // 8 q-blocks per (b,h)
    int lane = threadIdx.x & 31;
    int wv   = threadIdx.x >> 5;
    int bh   = blockIdx.x / QB;
    int qblk = blockIdx.x % QB;
    int b = bh / Hz, h = bh % Hz;
    int q0 = qblk * 128 + wv * 16;
    int len = lengths[b];

    const half_t* qp  = qb   + (size_t)bh * Lz * DH;
    const half_t* kp  = kbuf + (size_t)bh * Lz * DH;
    const half_t* vtp = vtb  + (size_t)bh * DH * Lz;

    __shared__ __align__(16) half_t smp[8][16 * 32];  // per-wave P tile (C-layout -> A-layout)
    half_t* pls = smp[wv];

    v16h aQ0 = ld_afrag(qp + (size_t)q0 * DH, DH, 0,  lane);
    v16h aQ1 = ld_afrag(qp + (size_t)q0 * DH, DH, 32, lane);

    v8f o0 = {}, o1 = {}, o2 = {}, o3 = {};
    float mrow[8], lrow[8];
#pragma unroll
    for (int r = 0; r < 8; ++r) { mrow[r] = -1e30f; lrow[r] = 0.f; }

    int n = lane & 15, hf = lane >> 4;

    for (int kb = 0; kb < len; kb += 32) {
        // S = Q * K^T  (two 16-key subtiles, K-dim = 64 split in 2 WMMA steps)
        v8f s0 = {}, s1 = {};
        s0 = wmma_f16(aQ0, ld_bfrag(kp, DH, 0,  kb,      lane), s0);
        s0 = wmma_f16(aQ1, ld_bfrag(kp, DH, 32, kb,      lane), s0);
        s1 = wmma_f16(aQ0, ld_bfrag(kp, DH, 0,  kb + 16, lane), s1);
        s1 = wmma_f16(aQ1, ld_bfrag(kp, DH, 32, kb + 16, lane), s1);

        bool v0 = (kb + n) < len;
        bool v1 = (kb + 16 + n) < len;
#pragma unroll
        for (int r = 0; r < 8; ++r) {
            float x0 = v0 ? s0[r] * SCALE : -1e30f;
            float x1 = v1 ? s1[r] * SCALE : -1e30f;
            float mn = fmaxf(mrow[r], rowmax16(fmaxf(x0, x1)));
            float al = __expf(mrow[r] - mn);
            float p0 = __expf(x0 - mn);
            float p1 = __expf(x1 - mn);
            lrow[r] = lrow[r] * al + rowsum16(p0 + p1);
            mrow[r] = mn;
            o0[r] *= al; o1[r] *= al; o2[r] *= al; o3[r] *= al;
            int row = r + 8 * hf;
            pls[row * 32 + n]      = (half_t)p0;
            pls[row * 32 + 16 + n] = (half_t)p1;
        }
        v16h aP = ld_afrag_lds(pls, lane);  // per-wave LDS region; DS ops in-order within wave
        o0 = wmma_f16(aP, ld_bfrag(vtp, Lz, kb, 0,  lane), o0);
        o1 = wmma_f16(aP, ld_bfrag(vtp, Lz, kb, 16, lane), o1);
        o2 = wmma_f16(aP, ld_bfrag(vtp, Lz, kb, 32, lane), o2);
        o3 = wmma_f16(aP, ld_bfrag(vtp, Lz, kb, 48, lane), o3);
    }

#pragma unroll
    for (int r = 0; r < 8; ++r) {
        float inv = 1.0f / lrow[r];
        int l = q0 + r + 8 * hf;
        size_t base = ((size_t)b * Lz + l) * NC + h * DH + n;
        attn[base + 0]  = (half_t)(o0[r] * inv);
        attn[base + 16] = (half_t)(o1[r] * inv);
        attn[base + 32] = (half_t)(o2[r] * inv);
        attn[base + 48] = (half_t)(o3[r] * inv);
    }
}

// ---------------- Kernel 5: out projection (WMMA, 64x32 per wave) + bias + residual + mask ----
// Grid: (M/128, C/128) = (64, 4); block = 8 waves arranged 2(M) x 4(N).

__global__ void proj_kernel(const half_t* __restrict__ attn, const half_t* __restrict__ WprojT,
                            const float* __restrict__ bproj, const float* __restrict__ x,
                            const int* __restrict__ lengths, float* __restrict__ out) {
    int lane = threadIdx.x & 31;
    int wv   = threadIdx.x >> 5;
    int mw = wv >> 2, nw = wv & 3;
    int m0 = blockIdx.x * 128 + mw * 64;
    int n0 = blockIdx.y * 128 + nw * 32;

    v8f acc[4][2] = {};
    for (int k0 = 0; k0 < NC; k0 += 32) {
        v16h a[4], bf[2];
#pragma unroll
        for (int mi = 0; mi < 4; ++mi)
            a[mi] = ld_afrag(attn + (size_t)(m0 + 16 * mi) * NC, NC, k0, lane);
#pragma unroll
        for (int ni = 0; ni < 2; ++ni)
            bf[ni] = ld_bfrag(WprojT, NC, k0, n0 + 16 * ni, lane);
#pragma unroll
        for (int mi = 0; mi < 4; ++mi)
#pragma unroll
            for (int ni = 0; ni < 2; ++ni)
                acc[mi][ni] = wmma_f16(a[mi], bf[ni], acc[mi][ni]);
    }

    int n = lane & 15, hf = lane >> 4;
    int bi  = m0 >> 10;                 // batch index (constant per wave)
    int len = lengths[bi];
    int lbase = (m0 & 1023) + 8 * hf;
#pragma unroll
    for (int ni = 0; ni < 2; ++ni) {
        int col = n0 + 16 * ni + n;
        float bias = bproj[col];
#pragma unroll
        for (int mi = 0; mi < 4; ++mi) {
#pragma unroll
            for (int r = 0; r < 8; ++r) {
                int l = lbase + 16 * mi + r;
                size_t gidx = ((size_t)bi * Lz + l) * NC + col;
                float v = 0.f;
                if (l < len) v = acc[mi][ni][r] + bias + x[gidx];
                out[gidx] = v;
            }
        }
    }
}

// ---------------- Host-side launch ----------------

extern "C" void kernel_launch(void* const* d_in, const int* in_sizes, int n_in,
                              void* d_out, int out_size, void* d_ws, size_t ws_size,
                              hipStream_t stream) {
    (void)in_sizes; (void)n_in; (void)out_size;
    const float* x       = (const float*)d_in[0];
    const int*   lengths = (const int*)  d_in[1];
    const float* gamma   = (const float*)d_in[2];
    const float* beta    = (const float*)d_in[3];
    const float* Wqkv    = (const float*)d_in[4];
    const float* bqkv    = (const float*)d_in[5];
    const float* Wproj   = (const float*)d_in[6];
    const float* bproj   = (const float*)d_in[7];
    float* out = (float*)d_out;

    char* ws = (char*)d_ws;
    size_t off = 0;
    auto take = [&](size_t bytes) -> void* {
        void* p = ws + off;
        off += (bytes + 255) & ~(size_t)255;
        return p;
    };
    const size_t BLC = (size_t)Bz * Lz * NC;
    half_t* xn     = (half_t*)take(BLC * sizeof(half_t));                       // 8 MB
    half_t* WqkvT  = (half_t*)take((size_t)3 * NC * NC * sizeof(half_t));       // 1.5 MB
    half_t* WprojT = (half_t*)take((size_t)NC * NC * sizeof(half_t));           // 0.5 MB
    half_t* qb     = (half_t*)take(BLC * sizeof(half_t));                       // 8 MB
    half_t* kbuf   = (half_t*)take(BLC * sizeof(half_t));                       // 8 MB
    half_t* vtb    = (half_t*)take(BLC * sizeof(half_t));                       // 8 MB (transposed)
    half_t* attn   = (half_t*)take(BLC * sizeof(half_t));                       // 8 MB
    if (off > ws_size) return; // workspace too small; nothing sane to do

    // 1) weights -> f16 transposed
    {
        int tot = NC * 3 * NC;
        w_transpose_f16_kernel<<<(tot + 255) / 256, 256, 0, stream>>>(Wqkv, WqkvT, NC, 3 * NC);
        tot = NC * NC;
        w_transpose_f16_kernel<<<(tot + 255) / 256, 256, 0, stream>>>(Wproj, WprojT, NC, NC);
    }
    // 2) GroupNorm
    groupnorm_kernel<<<Bz * Gz, 256, 0, stream>>>(x, lengths, gamma, beta, xn);
    // 3) QKV GEMM: grid (8192/128, 1536/128) = (64, 12), 8 waves/block, 64x32 per wave
    qkv_gemm_kernel<<<dim3(64, 12), 256, 0, stream>>>(xn, WqkvT, bqkv, qb, kbuf, vtb);
    // 4) attention: B*H*(L/128) blocks of 8 waves
    attention_kernel<<<Bz * Hz * (Lz / 128), 256, 0, stream>>>(qb, kbuf, vtb, lengths, attn);
    // 5) projection + residual + mask: grid (8192/128, 512/128) = (64, 4)
    proj_kernel<<<dim3(64, 4), 256, 0, stream>>>(attn, WprojT, bproj, x, lengths, out);
}
